// PerceiverAttn_14826227106229
// MI455X (gfx1250) — matrix operand order repaired and checked by reading, not compiled, not run on hardware
//
#include <hip/hip_runtime.h>
#include <hip/hip_bf16.h>

// ---------------------------------------------------------------------------
// PerceiverAttn on MI455X (gfx1250): bf16 WMMA GEMMs + f32 flash attention
// with Tensor-Data-Mover staged K/V tiles.
//   B=4, S=4096, E=1024, H=16, D=64, QL=512 (fixed per reference setup)
// ---------------------------------------------------------------------------

typedef __attribute__((ext_vector_type(16))) __bf16 v16bf;
typedef __attribute__((ext_vector_type(8)))  __bf16 v8bf;
typedef __attribute__((ext_vector_type(4)))  __bf16 v4bf;
typedef __attribute__((ext_vector_type(8)))  float  v8f;
typedef __attribute__((ext_vector_type(4)))  float  v4f;
typedef __attribute__((ext_vector_type(4)))  unsigned u32x4;
typedef __attribute__((ext_vector_type(4)))  int      i32x4;
typedef __attribute__((ext_vector_type(8)))  int      i32x8;

#define BB 4
#define SS 4096
#define EE 1024
#define HH 16
#define DD 64
#define QL 512
#define MASK_OFF (SS - QL - 1)   // keep key j iff j <= i + MASK_OFF

static __device__ inline v8f wmma_bf16(v16bf a, v16bf b, v8f c) {
  return __builtin_amdgcn_wmma_f32_16x16x32_bf16(
      /*neg_a=*/false, a, /*neg_b=*/false, b,
      /*c_mod=*/(short)0, c, /*reuse_a=*/false, /*reuse_b=*/false);
}

// v_permlane16_b32 lane-xor within 16-lane groups (pure VALU, no LDS pipe).
static __device__ inline float lane_perm(float v, unsigned lo, unsigned hi) {
  unsigned r = __builtin_amdgcn_permlane16(
      __builtin_bit_cast(unsigned, v), __builtin_bit_cast(unsigned, v),
      lo, hi, false, false);
  return __builtin_bit_cast(float, r);
}
#define X1LO 0x67452301u
#define X1HI 0xEFCDAB89u
#define X2LO 0x54761032u
#define X2HI 0xDCFE98BAu
#define X4LO 0x32107654u
#define X4HI 0xBA98FEDCu
#define X8LO 0xFEDCBA98u
#define X8HI 0x76543210u

static __device__ inline float red_max16(float m) {
  m = fmaxf(m, lane_perm(m, X1LO, X1HI));
  m = fmaxf(m, lane_perm(m, X2LO, X2HI));
  m = fmaxf(m, lane_perm(m, X4LO, X4HI));
  m = fmaxf(m, lane_perm(m, X8LO, X8HI));
  return m;
}
static __device__ inline float red_sum16(float s) {
  s += lane_perm(s, X1LO, X1HI);
  s += lane_perm(s, X2LO, X2HI);
  s += lane_perm(s, X4LO, X4HI);
  s += lane_perm(s, X8LO, X8HI);
  return s;
}

// A-matrix 16x32 bf16 fragment (this file's k order, B matches; not the ISA table): lane = row (lane%16); the 16
// per-lane elements are k = g*16..+15 contiguous  (g = lane/16).
static __device__ inline v16bf load_a_frag(const __bf16* p, int g) {
  v8bf lo = *(const v8bf*)(p + g * 16);
  v8bf hi = *(const v8bf*)(p + g * 16 + 8);
  return __builtin_shufflevector(lo, hi, 0, 1, 2, 3, 4, 5, 6, 7,
                                 8, 9, 10, 11, 12, 13, 14, 15);
}

// --------------------- Tensor Data Mover (TDM) helpers ---------------------
// D# group0: [1:0]=count(1), [63:32]=lds_addr, [120:64]=global_addr,
//            [127:126]=type(2).  All words forced uniform -> SGPR tuple.
static __device__ inline u32x4 tdm_g0(unsigned lds_off, unsigned long long ga) {
  u32x4 g;
  g[0] = 1u;
  g[1] = (unsigned)__builtin_amdgcn_readfirstlane((int)lds_off);
  g[2] = (unsigned)__builtin_amdgcn_readfirstlane((int)(unsigned)ga);
  g[3] = ((unsigned)__builtin_amdgcn_readfirstlane((int)(unsigned)(ga >> 32)) &
          0x01FFFFFFu) | 0x80000000u;   // type=2 ("image")
  return g;
}

static __device__ inline void tdm_load(u32x4 g0, i32x8 g1) {
  i32x4 z4 = {};
#if __clang_major__ >= 23
  i32x8 z8 = {};
  __builtin_amdgcn_tensor_load_to_lds(g0, g1, z4, z4, z8, 0);
#else
  __builtin_amdgcn_tensor_load_to_lds(g0, g1, z4, z4, 0);
#endif
}

// ---------------------------------------------------------------------------
// fp32 -> bf16 conversion, 4 elements per thread
// ---------------------------------------------------------------------------
__global__ void cvt_bf16(const v4f* __restrict__ src, v4bf* __restrict__ dst,
                         int n4) {
  int i = blockIdx.x * blockDim.x + threadIdx.x;
  int stride = gridDim.x * blockDim.x;
  for (; i < n4; i += stride) {
    v4f s = src[i];
    v4bf d;
    d[0] = (__bf16)s[0]; d[1] = (__bf16)s[1];
    d[2] = (__bf16)s[2]; d[3] = (__bf16)s[3];
    dst[i] = d;
  }
}

// ---------------------------------------------------------------------------
// Projection GEMM y = x @ W^T, one wave per 32(M)x64(N) tile (N == one head),
// 8 WMMAs per k-step; fused rotary + layout scatter.
//   MODE 0: K -> rotary(pos=s),        out[b,h,s,d]
//   MODE 1: V -> no rotary,            out[b,h,d,s]  (transposed for PV GEMM)
//   MODE 2: Q -> rotary(pos=i)*0.125,  out[b,h,i,d]  (1/sqrt(d) folded in)
// ---------------------------------------------------------------------------
template <int MODE>
__global__ __launch_bounds__(128) void proj_rope(const __bf16* __restrict__ X,
                                                 const __bf16* __restrict__ W,
                                                 __bf16* __restrict__ out) {
  const int lane = threadIdx.x & 31;
  const int warp = threadIdx.x >> 5;
  const int wid  = blockIdx.x * 4 + warp;
  const int mt   = wid >> 4;
  const int nt   = wid & 15;
  const int col  = lane & 15;
  const int grp  = lane >> 4;

  const int ROWS_PER_B = (MODE == 2) ? QL : SS;

  const __bf16* xr[2];
#pragma unroll
  for (int mi = 0; mi < 2; ++mi) {
    const int arow = mt * 32 + mi * 16 + col;
    const int ab   = arow / ROWS_PER_B;
    const int apos = arow % ROWS_PER_B;
    const int xrow = (MODE == 2) ? (ab * SS + (SS - QL) + apos) : arow;
    xr[mi] = X + (size_t)xrow * EE;
  }

  v8f acc[2][4] = {};

#pragma unroll 2
  for (int kk = 0; kk < EE; kk += 32) {
    v16bf a0 = load_a_frag(xr[0] + kk, grp);
    v16bf a1 = load_a_frag(xr[1] + kk, grp);
#pragma unroll
    for (int f = 0; f < 4; ++f) {
      const __bf16* wr = W + (size_t)(nt * 64 + f * 16 + col) * EE + kk + grp * 16;
      v16bf bf = *(const v16bf*)wr;
      acc[0][f] = wmma_bf16(a0, bf, acc[0][f]);
      acc[1][f] = wmma_bf16(a1, bf, acc[1][f]);
    }
  }

#pragma unroll
  for (int f = 0; f < 4; ++f) {
    const int d = f * 16 + col;
    const float inv = __expf((float)(d >> 1) * -0.2878231366242592f);
#pragma unroll
    for (int mi = 0; mi < 2; ++mi) {
#pragma unroll
      for (int r = 0; r < 8; ++r) {
        const int m   = mt * 32 + mi * 16 + grp * 8 + r;
        const int b   = m / ROWS_PER_B;
        const int pos = m % ROWS_PER_B;
        float val = acc[mi][f][r];
        if (MODE != 1) {
          float sn, cs;
          __sincosf((float)pos * inv, &sn, &cs);
          const float partner = lane_perm(val, X1LO, X1HI);
          val = val * cs + ((d & 1) ? partner : -partner) * sn;
          if (MODE == 2) val *= 0.125f;
        }
        size_t idx;
        if (MODE == 0)      idx = ((size_t)(b * HH + nt) * SS + pos) * DD + d;
        else if (MODE == 1) idx = ((size_t)(b * HH + nt) * DD + d) * SS + pos;
        else                idx = ((size_t)(b * HH + nt) * QL + pos) * DD + d;
        out[idx] = (__bf16)val;
      }
    }
  }
}

// ---------------------------------------------------------------------------
// Flash attention with TDM-staged tiles: one wave per (b, h, 16-query tile).
// K tiles (32 keys x 64 d, 4KB) double-buffered in LDS via tensor_load_to_lds;
// V tiles (64 d x 32 keys from transposed V) single-buffered (issued at the
// top of the iteration, consumed after the softmax stretch).
// TENSORcnt discipline per chunk n (issue order ... K(n), V(n), K(n+1)):
//   s_wait_tensorcnt 2 before score WMMAs  -> K(n) resident
//   s_wait_tensorcnt 1 before PV WMMAs     -> V(n) resident
// ---------------------------------------------------------------------------
__global__ __launch_bounds__(128) void flash_attn(const __bf16* __restrict__ Qb,
                                                  const __bf16* __restrict__ Kb,
                                                  const __bf16* __restrict__ Vt,
                                                  float* __restrict__ out) {
  __shared__ __attribute__((aligned(64))) __bf16 kbuf[4][2][32 * 64]; // 32 KB
  __shared__ __attribute__((aligned(64))) __bf16 vbuf[4][64 * 32];    // 16 KB
  __shared__ __attribute__((aligned(64))) __bf16 plds[4][16 * 32];    //  4 KB

  const int lane = threadIdx.x & 31;
  const int warp = threadIdx.x >> 5;
  const int wid  = blockIdx.x * 4 + warp;
  const int qt   = wid & 31;
  const int h    = (wid >> 5) & 15;
  const int b    = wid >> 9;
  const int col  = lane & 15;
  const int grp  = lane >> 4;
  const int qbase = qt * 16;

  // Q A-fragments (pre-scaled by 0.125 in proj_rope<2>)
  const __bf16* qr = Qb + ((size_t)(b * HH + h) * QL + qbase + col) * DD;
  const v16bf aq0 = load_a_frag(qr, grp);
  const v16bf aq1 = load_a_frag(qr + 32, grp);

  const __bf16* kbase = Kb + (size_t)(b * HH + h) * SS * DD;  // [s][d]
  const __bf16* vbase = Vt + (size_t)(b * HH + h) * DD * SS;  // [d][s]
  __bf16* pl = &plds[warp][0];

  // TDM descriptors: group1 bit-packing per ISA 8.4 (data_size=2B)
  // K tensor [4096 s][64 d]: dim0=64, dim0_stride=64, dim1=4096, tile 64x32
  const i32x8 g1k = {0x00010000, 64 << 16, 4096 << 16, 64 << 16, 32, 64, 0, 0};
  // V tensor [64 d][4096 s]: dim0=4096, dim0_stride=4096, dim1=64, tile 32x64
  const i32x8 g1v = {0x00010000, 4096 << 16, 64 << 16, 32 << 16, 64, 4096, 0, 0};

  const unsigned long long ka = (unsigned long long)(uintptr_t)kbase;
  const unsigned long long va = (unsigned long long)(uintptr_t)vbase;
  const unsigned kofs0 = (unsigned)(uintptr_t)&kbuf[warp][0][0];
  const unsigned kofs1 = (unsigned)(uintptr_t)&kbuf[warp][1][0];
  const unsigned vofs  = (unsigned)(uintptr_t)&vbuf[warp][0];

  float mrun[8], lrun[8];
#pragma unroll
  for (int r = 0; r < 8; ++r) { mrun[r] = -3.0e38f; lrun[r] = 0.f; }
  v8f o[4] = {};

  int kend = qbase + 15 + MASK_OFF + 1;
  if (kend > SS) kend = SS;
  kend = (kend + 31) & ~31;
  int kfull = (((qbase + MASK_OFF - 31) >> 5) << 5) + 32;
  if (kfull > kend) kfull = kend;

  // prime the pipeline: K tile for chunk 0
  tdm_load(tdm_g0(kofs0, ka), g1k);

  const int nchunks = kend >> 5;
  for (int it = 0; it < nchunks; ++it) {
    const int kc  = it << 5;
    const int cur = it & 1;

    // issue V(kc) and K(kc+32) tile DMAs
    tdm_load(tdm_g0(vofs, va + (unsigned long long)kc * 2), g1v);
    int knc = kc + 32;
    if (knc > SS - 32) knc = SS - 32;
    tdm_load(tdm_g0(cur ? kofs0 : kofs1,
                    ka + (unsigned long long)knc * (DD * 2)), g1k);

    __builtin_amdgcn_s_wait_tensorcnt(2);   // K(kc) resident
    asm volatile("" ::: "memory");

    // K B-fragments from LDS tile [32 keys][64 d]
    const __bf16* kb = &kbuf[warp][cur][0];
    v16bf k00 = *(const v16bf*)(kb + (size_t)col * 64 + grp * 16);
    v16bf k01 = *(const v16bf*)(kb + (size_t)col * 64 + 32 + grp * 16);
    v16bf k10 = *(const v16bf*)(kb + (size_t)(16 + col) * 64 + grp * 16);
    v16bf k11 = *(const v16bf*)(kb + (size_t)(16 + col) * 64 + 32 + grp * 16);

    v8f sc0 = {}, sc1 = {};
    sc0 = wmma_bf16(aq0, k00, sc0);
    sc0 = wmma_bf16(aq1, k01, sc0);
    sc1 = wmma_bf16(aq0, k10, sc1);
    sc1 = wmma_bf16(aq1, k11, sc1);

    const bool masked = (kc >= kfull);   // uniform branch, boundary chunks only

    // online softmax (row stats across the owning 16-lane half, pure VALU)
#pragma unroll
    for (int r = 0; r < 8; ++r) {
      float s0 = sc0[r], s1 = sc1[r];
      if (masked) {
        const int lim = qbase + grp * 8 + r + MASK_OFF;
        s0 = (kc + col      <= lim) ? s0 : -3.0e38f;
        s1 = (kc + 16 + col <= lim) ? s1 : -3.0e38f;
      }
      const float m     = red_max16(fmaxf(s0, s1));
      const float mnew  = fmaxf(mrun[r], m);
      const float alpha = __expf(mrun[r] - mnew);
      mrun[r] = mnew;
      const float p0 = __expf(s0 - mnew);
      const float p1 = __expf(s1 - mnew);
      lrun[r] = lrun[r] * alpha + red_sum16(p0 + p1);
      o[0][r] *= alpha; o[1][r] *= alpha; o[2][r] *= alpha; o[3][r] *= alpha;
      const int rl = grp * 8 + r;
      pl[rl * 32 + col]      = (__bf16)p0;
      pl[rl * 32 + 16 + col] = (__bf16)p1;
    }

    asm volatile("s_wait_dscnt 0" ::: "memory");  // P stores -> P reads
    v16bf pa = load_a_frag(pl + col * 32, grp);

    __builtin_amdgcn_s_wait_tensorcnt(1);   // V(kc) resident
    asm volatile("" ::: "memory");

    // V B-fragments from LDS tile [64 d][32 keys]
    const __bf16* vb = &vbuf[warp][0];
#pragma unroll
    for (int f = 0; f < 4; ++f) {
      v16bf bv = *(const v16bf*)(vb + (size_t)(f * 16 + col) * 32 + grp * 16);
      o[f] = wmma_bf16(pa, bv, o[f]);
    }
  }

  // normalize and write out[b, i, h*64 + d] (fp32)
  float* ob = out + (size_t)b * QL * EE + (size_t)h * DD;
#pragma unroll
  for (int f = 0; f < 4; ++f) {
    const int d = f * 16 + col;
#pragma unroll
    for (int r = 0; r < 8; ++r) {
      const int i = qbase + grp * 8 + r;
      ob[(size_t)i * EE + d] = o[f][r] / lrun[r];
    }
  }
}

// ---------------------------------------------------------------------------
extern "C" void kernel_launch(void* const* d_in, const int* in_sizes, int n_in,
                              void* d_out, int out_size, void* d_ws,
                              size_t ws_size, hipStream_t stream) {
  (void)in_sizes; (void)n_in; (void)out_size; (void)ws_size;

  const float* x  = (const float*)d_in[0];
  const float* Wq = (const float*)d_in[1];
  const float* Wk = (const float*)d_in[2];
  const float* Wv = (const float*)d_in[3];
  // d_in[4] = q_len; fixed at 512 per the reference setup (compile-time).

  char* ws = (char*)d_ws;
  __bf16* x_bf  = (__bf16*)(ws);               // 32 MiB
  __bf16* wq_bf = (__bf16*)(ws + 33554432);    // 2 MiB
  __bf16* wk_bf = (__bf16*)(ws + 35651584);    // 2 MiB
  __bf16* wv_bf = (__bf16*)(ws + 37748736);    // 2 MiB
  __bf16* k_bf  = (__bf16*)(ws + 39845888);    // [b,h,s,d]   32 MiB
  __bf16* v_t   = (__bf16*)(ws + 73400320);    // [b,h,d,s]   32 MiB
  __bf16* q_bf  = (__bf16*)(ws + 106954752);   // [b,h,i,d]    4 MiB

  cvt_bf16<<<2048, 256, 0, stream>>>((const v4f*)x,  (v4bf*)x_bf,  BB * SS * EE / 4);
  cvt_bf16<<<512,  256, 0, stream>>>((const v4f*)Wq, (v4bf*)wq_bf, EE * EE / 4);
  cvt_bf16<<<512,  256, 0, stream>>>((const v4f*)Wk, (v4bf*)wk_bf, EE * EE / 4);
  cvt_bf16<<<512,  256, 0, stream>>>((const v4f*)Wv, (v4bf*)wv_bf, EE * EE / 4);

  proj_rope<0><<<2048, 128, 0, stream>>>(x_bf, wk_bf, k_bf);
  proj_rope<1><<<2048, 128, 0, stream>>>(x_bf, wv_bf, v_t);
  proj_rope<2><<<256, 128, 0, stream>>>(x_bf, wq_bf, q_bf);

  flash_attn<<<512, 128, 0, stream>>>(q_bf, k_bf, v_t, (float*)d_out);
}
